// LlamaAttention_16423954940491
// MI455X (gfx1250) — matrix operand order repaired and checked by reading, not compiled, not run on hardware
//
#include <hip/hip_runtime.h>
#include <hip/hip_bf16.h>

// ---------------------------------------------------------------------------
// LlamaAttention for MI455X (gfx1250): f16 WMMA GEMMs + fused LDS softmax.
// ---------------------------------------------------------------------------

#define S_   2048
#define HID_ 2048
#define NH_  32
#define NKV_ 8
#define HD_  64
#define NEG_ (-1.0e9f)

typedef _Float16 v8h  __attribute__((ext_vector_type(8)));
typedef _Float16 v16h __attribute__((ext_vector_type(16)));
typedef float    v8f  __attribute__((ext_vector_type(8)));

static __device__ inline v8f wmma_f16(v16h a, v16h b, v8f c) {
  // D = A(16x32 f16) * B(32x16 f16) + C(16x16 f32)
  return __builtin_amdgcn_wmma_f32_16x16x32_f16(false, a, false, b,
                                                (short)0, c, false, false);
}

// A-matrix 16-bit 16x32 fragment (ISA 7.12.2):
//  lanes 0-15 (M=lane):   h[0..7]=K 0..7,  h[8..15]=K16..23
//  lanes16-31 (M=lane-16):h[0..7]=K 8..15, h[8..15]=K24..31
static __device__ inline v16h load_a_frag(const _Float16* p /* row + kbase + (lane&16?8:0) */) {
  v8h lo = *(const v8h*)(p);
  v8h hi = *(const v8h*)(p + 16);
  v16h r;
#pragma unroll
  for (int j = 0; j < 8; ++j) { r[j] = lo[j]; r[8 + j] = hi[j]; }
  return r;
}

// ---------------------------------------------------------------------------
// f32 -> f16 cast
// ---------------------------------------------------------------------------
__global__ void __launch_bounds__(256)
cast_f16_kernel(const float* __restrict__ src, _Float16* __restrict__ dst, int n) {
  int i = blockIdx.x * 256 + threadIdx.x;
  if (i < n) dst[i] = (_Float16)src[i];
}

// ---------------------------------------------------------------------------
// TN GEMM: C(MxN) = A(MxK, row-major f16) * B(NxK, row-major f16)^T
// mode 0: C -> f16 head-packed   (head=col/64, s=row, d=col%64), head stride M
// mode 1: C -> f32 row-major (M x N)
// mode 2: C -> f16 head-packed TRANSPOSED: (head, d, s)  [for V]
// Block: 256 threads = 8 waves; WG tile 64x128; each wave computes 16x64
// (one A fragment feeds 4 WMMAs -> 2x arithmetic intensity vs 16x32 tiles).
// ---------------------------------------------------------------------------
__global__ void __launch_bounds__(256)
gemm_tn_kernel(const _Float16* __restrict__ A, const _Float16* __restrict__ B,
               void* __restrict__ C, int M, int N, int K, int mode) {
  const int t = threadIdx.x;
  const int lane = t & 31;
  const int wave = t >> 5;
  const int waveM = wave >> 1;           // 0..3
  const int waveN = wave & 1;            // 0..1
  const int row_base = blockIdx.y * 64 + waveM * 16;
  const int col_base = blockIdx.x * 128 + waveN * 64;

  const int m    = lane & 15;
  const int aoff = (lane & 16) ? 16 : 0;
  const int boff = (lane & 16) ? 16 : 0;

  const _Float16* arow = A + (size_t)(row_base + m) * K + aoff;
  const _Float16* bp   = B + (size_t)(col_base + m) * K + boff;

  v8f c0 = {}, c1 = {}, c2 = {}, c3 = {};
  for (int kb = 0; kb < K; kb += 32) {
    v16h a  = *(const v16h*)(arow + kb);
    v16h b0 = *(const v16h*)(bp + kb);                    // 16 contiguous K halves
    v16h b1 = *(const v16h*)(bp + (size_t)16 * K + kb);
    v16h b2 = *(const v16h*)(bp + (size_t)32 * K + kb);
    v16h b3 = *(const v16h*)(bp + (size_t)48 * K + kb);
    c0 = wmma_f16(a, b0, c0);
    c1 = wmma_f16(a, b1, c1);
    c2 = wmma_f16(a, b2, c2);
    c3 = wmma_f16(a, b3, c3);
  }

  const int rbase = (lane & 16) ? 8 : 0;
  const int ccol  = lane & 15;
  v8f acc[4] = {c0, c1, c2, c3};
  if (mode == 0) {
    _Float16* O = (_Float16*)C;
#pragma unroll
    for (int j = 0; j < 4; ++j) {
      int col = col_base + j * 16 + ccol;
      size_t base = ((size_t)(col >> 6) * M) * 64 + (col & 63);
#pragma unroll
      for (int i = 0; i < 8; ++i) {
        int row = row_base + rbase + i;
        O[base + (size_t)row * 64] = (_Float16)acc[j][i];
      }
    }
  } else if (mode == 1) {
    float* O = (float*)C;
#pragma unroll
    for (int j = 0; j < 4; ++j) {
      int col = col_base + j * 16 + ccol;
#pragma unroll
      for (int i = 0; i < 8; ++i) {
        int row = row_base + rbase + i;
        O[(size_t)row * N + col] = acc[j][i];
      }
    }
  } else {  // mode 2: transposed (head, d, s) pack for V
    _Float16* O = (_Float16*)C;
#pragma unroll
    for (int j = 0; j < 4; ++j) {
      int col = col_base + j * 16 + ccol;
      size_t base = ((size_t)(col >> 6) * HD_ + (col & 63)) * (size_t)M;
#pragma unroll
      for (int i = 0; i < 8; ++i) {
        int row = row_base + rbase + i;
        O[base + row] = (_Float16)acc[j][i];
      }
    }
  }
}

// ---------------------------------------------------------------------------
// In-place RoPE on (heads, S, HD) f16; one thread owns the (d, d+32) pair.
// ---------------------------------------------------------------------------
__global__ void __launch_bounds__(256)
rope_kernel(_Float16* __restrict__ X, const float* __restrict__ cosp,
            const float* __restrict__ sinp, int nheads) {
  int idx = blockIdx.x * 256 + threadIdx.x;
  int total = nheads * S_ * 32;
  if (idx >= total) return;
  int d = idx & 31;
  int r = (idx >> 5) & (S_ - 1);
  int h = idx >> 16;                      // 32*S_ = 65536
  size_t base = ((size_t)h * S_ + r) * 64;
  float c  = cosp[r * 64 + d];
  float sn = sinp[r * 64 + d];
  float x0 = (float)X[base + d];
  float x1 = (float)X[base + d + 32];
  X[base + d]      = (_Float16)(x0 * c - x1 * sn);
  X[base + d + 32] = (_Float16)(x1 * c + x0 * sn);
}

// ---------------------------------------------------------------------------
// Fused attention: one WG per (head, 16-query tile).
// Full 16x2048 fp32 score row in LDS (128 KB of the 320 KB WGP LDS).
// 8 waves each own a 256-key slice for QK^T and P*V WMMAs.
// V is pre-transposed (head, d, s) so P*V B-fragments are contiguous loads.
// ---------------------------------------------------------------------------
__global__ void __launch_bounds__(256)
attn_kernel(const _Float16* __restrict__ Qh, const _Float16* __restrict__ Kh,
            const _Float16* __restrict__ Vt, float* __restrict__ probs,
            _Float16* __restrict__ AOh) {
  __shared__ float    sc[16][S_];        // 128 KB scores / probs / O-partials
  __shared__ _Float16 qt[16 * 64];       // 2 KB Q tile
  __shared__ float    red[16][16];
  __shared__ float    rowstat[16];

  const int t    = threadIdx.x;
  const int lane = t & 31;
  const int wave = t >> 5;
  const int qtile = blockIdx.x;
  const int head  = blockIdx.y;
  const int hkv   = head >> 2;           // NH/NKV = 4 (GQA)
  const int q0    = qtile * 16;

  for (int i = t; i < 16 * 64; i += 256) {
    int r = i >> 6, d = i & 63;
    qt[i] = Qh[((size_t)head * S_ + q0 + r) * 64 + d];
  }
  __syncthreads();

  // Q A-fragments (reused across all key tiles by this wave).
  v16h aq0, aq1;
  {
    int m   = lane & 15;
    int off = (lane & 16) ? 16 : 0;
    const _Float16* p0 = &qt[m * 64 + off];
    const _Float16* p1 = &qt[m * 64 + 32 + off];
#pragma unroll
    for (int j = 0; j < 8; ++j) {
      aq0[j] = p0[j]; aq0[8 + j] = p0[8 + j];
      aq1[j] = p1[j]; aq1[8 + j] = p1[8 + j];
    }
  }

  // --- QK^T over this wave's 256-key slice ---
  const float scaling = 0.125f;          // HD^-0.5
  for (int nt = 0; nt < 16; ++nt) {
    int keyb = wave * 256 + nt * 16;
    int key  = keyb + (lane & 15);
    int doff = (lane & 16) ? 16 : 0;
    const _Float16* kp = Kh + ((size_t)hkv * S_ + key) * 64 + doff;
    __builtin_prefetch(kp + 16 * 64, 0, 0);   // next key tile -> global_prefetch
    v16h b0 = *(const v16h*)(kp);
    v16h b1 = *(const v16h*)(kp + 32);
    v8f c = {};
    c = wmma_f16(aq0, b0, c);
    c = wmma_f16(aq1, b1, c);
    int rbase = (lane & 16) ? 8 : 0;
    int col   = keyb + (lane & 15);
#pragma unroll
    for (int i = 0; i < 8; ++i) {
      int row = rbase + i;
      sc[row][col] = c[i] * scaling + ((col <= q0 + row) ? 0.0f : NEG_);
    }
  }
  __syncthreads();

  // --- Softmax: 16 threads per 2048-wide row ---
  {
    int r = t >> 4, cidx = t & 15;
    float mx = -3.0e38f;
    for (int col = cidx; col < S_; col += 16) mx = fmaxf(mx, sc[r][col]);
    red[r][cidx] = mx;
  }
  __syncthreads();
  if ((t & 15) == 0) {
    int r = t >> 4;
    float mx = red[r][0];
#pragma unroll
    for (int j = 1; j < 16; ++j) mx = fmaxf(mx, red[r][j]);
    rowstat[r] = mx;
  }
  __syncthreads();
  {
    int r = t >> 4, cidx = t & 15;
    float mx = rowstat[r];
    float s = 0.0f;
    for (int col = cidx; col < S_; col += 16) {
      float e = __expf(sc[r][col] - mx);
      sc[r][col] = e;
      s += e;
    }
    red[r][cidx] = s;
  }
  __syncthreads();
  if ((t & 15) == 0) {
    int r = t >> 4;
    float s = 0.0f;
#pragma unroll
    for (int j = 0; j < 16; ++j) s += red[r][j];
    rowstat[r] = 1.0f / s;
  }
  __syncthreads();

  // Normalize in LDS and stream attn_weights to HBM (coalesced).
  float* pout = probs + ((size_t)head * S_ + q0) * S_;
  for (int i = t; i < 16 * S_; i += 256) {
    int r = i >> 11, col = i & (S_ - 1);
    float p = sc[r][col] * rowstat[r];
    sc[r][col] = p;
    pout[(size_t)r * S_ + col] = p;
  }
  __syncthreads();

  // --- P * V over this wave's 256-key slice (contiguous B loads via Vt) ---
  v8f o0 = {}, o1 = {}, o2 = {}, o3 = {};
  {
    int m    = lane & 15;
    int aoff = (lane & 16) ? 16 : 0;
    int koff = (lane & 16) ? 16 : 0;
    const _Float16* vrow = Vt + ((size_t)hkv * HD_ + (lane & 15)) * S_ + koff;
    for (int kb = 0; kb < 8; ++kb) {
      int key0 = wave * 256 + kb * 32;
      v16h ap;
#pragma unroll
      for (int j = 0; j < 8; ++j) {
        ap[j]     = (_Float16)sc[m][key0 + aoff + j];
        ap[8 + j] = (_Float16)sc[m][key0 + aoff + 8 + j];
      }
      const _Float16* vb = vrow + key0;
      v16h b0 = *(const v16h*)(vb);                       // d = lane&15
      v16h b1 = *(const v16h*)(vb + (size_t)16 * S_);     // d = 16 + lane&15
      v16h b2 = *(const v16h*)(vb + (size_t)32 * S_);
      v16h b3 = *(const v16h*)(vb + (size_t)48 * S_);
      o0 = wmma_f16(ap, b0, o0);
      o1 = wmma_f16(ap, b1, o1);
      o2 = wmma_f16(ap, b2, o2);
      o3 = wmma_f16(ap, b3, o3);
    }
  }
  __syncthreads();  // all waves done reading sc: reuse it for O partials

  // Deterministic cross-wave reduction (no float atomics).
  {
    float* part = &sc[0][0] + wave * 1024;   // [wave][row][64]
    int rbase = (lane & 16) ? 8 : 0;
    int ccol  = lane & 15;
#pragma unroll
    for (int i = 0; i < 8; ++i) {
      int row = rbase + i;
      part[row * 64 + ccol]      = o0[i];
      part[row * 64 + 16 + ccol] = o1[i];
      part[row * 64 + 32 + ccol] = o2[i];
      part[row * 64 + 48 + ccol] = o3[i];
    }
  }
  __syncthreads();
  for (int i = t; i < 16 * 64; i += 256) {
    int r = i >> 6, d = i & 63;
    float acc = 0.0f;
#pragma unroll
    for (int w = 0; w < 8; ++w) acc += (&sc[0][0])[w * 1024 + r * 64 + d];
    AOh[(size_t)(q0 + r) * (NH_ * HD_) + head * 64 + d] = (_Float16)acc;
  }
}

// ---------------------------------------------------------------------------
// Launch
// ---------------------------------------------------------------------------
extern "C" void kernel_launch(void* const* d_in, const int* in_sizes, int n_in,
                              void* d_out, int out_size, void* d_ws, size_t ws_size,
                              hipStream_t stream) {
  (void)in_sizes; (void)n_in; (void)out_size; (void)ws_size;
  const float* hs   = (const float*)d_in[0];
  const float* cosp = (const float*)d_in[1];
  const float* sinp = (const float*)d_in[2];
  // d_in[3] = attention_mask: analytic causal mask, recomputed from indices
  const float* Wq = (const float*)d_in[4];
  const float* Wk = (const float*)d_in[5];
  const float* Wv = (const float*)d_in[6];
  const float* Wo = (const float*)d_in[7];

  char* w = (char*)d_ws;                       // 48 MiB of f16 staging
  _Float16* Xh  = (_Float16*)(w);              //  8 MiB  (S x HID)
  _Float16* Wqh = (_Float16*)(w + (size_t)( 8u << 20));
  _Float16* Wkh = (_Float16*)(w + (size_t)(16u << 20));
  _Float16* Wvh = (_Float16*)(w + (size_t)(18u << 20));
  _Float16* Woh = (_Float16*)(w + (size_t)(20u << 20));
  _Float16* Qh  = (_Float16*)(w + (size_t)(28u << 20));  // (NH, S, HD)
  _Float16* Kh  = (_Float16*)(w + (size_t)(36u << 20));  // (NKV, S, HD)
  _Float16* Vt  = (_Float16*)(w + (size_t)(38u << 20));  // (NKV, HD, S)  transposed
  _Float16* AOh = (_Float16*)(w + (size_t)(40u << 20));  // (S, NH*HD)

  cast_f16_kernel<<<(S_ * HID_) / 256, 256, 0, stream>>>(hs, Xh, S_ * HID_);
  cast_f16_kernel<<<(NH_ * HD_ * HID_) / 256, 256, 0, stream>>>(Wq, Wqh, NH_ * HD_ * HID_);
  cast_f16_kernel<<<(NKV_ * HD_ * HID_) / 256, 256, 0, stream>>>(Wk, Wkh, NKV_ * HD_ * HID_);
  cast_f16_kernel<<<(NKV_ * HD_ * HID_) / 256, 256, 0, stream>>>(Wv, Wvh, NKV_ * HD_ * HID_);
  cast_f16_kernel<<<(HID_ * NH_ * HD_) / 256, 256, 0, stream>>>(Wo, Woh, HID_ * NH_ * HD_);

  // Q/K projections: head-packed f16.  V projection: transposed head pack.
  gemm_tn_kernel<<<dim3((NH_ * HD_) / 128, S_ / 64), 256, 0, stream>>>(
      Xh, Wqh, Qh, S_, NH_ * HD_, HID_, 0);
  gemm_tn_kernel<<<dim3((NKV_ * HD_) / 128, S_ / 64), 256, 0, stream>>>(
      Xh, Wkh, Kh, S_, NKV_ * HD_, HID_, 0);
  gemm_tn_kernel<<<dim3((NKV_ * HD_) / 128, S_ / 64), 256, 0, stream>>>(
      Xh, Wvh, Vt, S_, NKV_ * HD_, HID_, 2);

  rope_kernel<<<(NH_ * S_ * 32) / 256, 256, 0, stream>>>(Qh, cosp, sinp, NH_);
  rope_kernel<<<(NKV_ * S_ * 32) / 256, 256, 0, stream>>>(Kh, cosp, sinp, NKV_);

  float* out = (float*)d_out;                  // [attn_output | attn_weights]
  attn_kernel<<<dim3(S_ / 16, NH_), 256, 0, stream>>>(
      Qh, Kh, Vt, out + (size_t)S_ * HID_, AOh);

  // Output projection -> fp32 attn_output
  gemm_tn_kernel<<<dim3(HID_ / 128, S_ / 64), 256, 0, stream>>>(
      AOh, Woh, out, S_, HID_, NH_ * HD_, 1);
}
